// SAM_1769526525971
// MI455X (gfx1250) — matrix operand order repaired and checked by reading, not compiled, not run on hardware
//
#include <hip/hip_runtime.h>
#include <cstdint>

#define CIN_ 512
#define HW_ 1024
#define MTOT_ 640
#define KK_ 49
#define PIX_ 8
#define NCOL_ 400      // 49*8 = 392 padded to 400 (25 N-tiles of 16)
#define CP_ 96         // 66 channels padded to 96 (3 K-steps of 32)
#define TT_STRIDE 104  // bf16 elems per col in tT  (16B aligned: 208B)
#define T2_STRIDE 72   // bf16 elems per col in t2T (16B aligned: 144B)
#define WGT_STRIDE 68  // f32 elems per col in wgt

// LDS byte offsets inside san_fused
#define SM_T2_OFF   (NCOL_ * WGT_STRIDE * 4)                  // 108800
#define SM_CW1_OFF  (SM_T2_OFF + NCOL_ * T2_STRIDE * 2)       // 166400
#define SM_CW2_OFF  (SM_CW1_OFF + 64 * CP_ * 2)               // 178688
#define SM_TOTAL    (SM_CW2_OFF + 64 * 64 * 2)                // 186880

typedef __bf16 v16bf __attribute__((ext_vector_type(16)));
typedef __bf16 v8bf  __attribute__((ext_vector_type(8)));
typedef float  v8f   __attribute__((ext_vector_type(8)));

__device__ __forceinline__ __bf16 f2bf(float f) {
  unsigned u = __float_as_uint(f);
  unsigned r = (u + 0x7FFFu + ((u >> 16) & 1u)) >> 16;   // round-to-nearest-even
  unsigned short h = (unsigned short)r;
  return __builtin_bit_cast(__bf16, h);
}

__device__ __forceinline__ int refl32(int i) {   // jnp 'reflect' for size 32, pad<=3
  if (i < 0) i = -i;
  if (i >= 32) i = 62 - i;
  return i;
}

// ---------------------------------------------------------------------------
// Prep 1: x [B,512,32,32] f32 -> xbT [B][HW][CIN] bf16 (transposed for WMMA-B)
// ---------------------------------------------------------------------------
__global__ __launch_bounds__(256) void k_prep_x(const float* __restrict__ x,
                                                __bf16* __restrict__ xbT) {
  int idx = blockIdx.x * 256 + threadIdx.x;       // = (b*HW + n)*CIN + c
  if (idx >= 4 * HW_ * CIN_) return;
  int c = idx & (CIN_ - 1);
  int t = idx / CIN_;
  int n = t & (HW_ - 1);
  int b = t / HW_;
  xbT[idx] = f2bf(x[((size_t)b * CIN_ + c) * HW_ + n]);
}

// ---------------------------------------------------------------------------
// Prep 2: weights -> bf16 (wcat [640][512], wcw1b [64][96] zero-padded,
// wcw2b [64][64]), biascat [640], BN scale/shift (bnp: s0[66] h0[66] s1[64] h1[64])
// ---------------------------------------------------------------------------
__global__ __launch_bounds__(256) void k_prep_w(
    const float* __restrict__ w1, const float* __restrict__ b1,
    const float* __restrict__ w2, const float* __restrict__ b2,
    const float* __restrict__ w3, const float* __restrict__ b3,
    const float* __restrict__ g0, const float* __restrict__ be0,
    const float* __restrict__ m0, const float* __restrict__ v0,
    const float* __restrict__ wcw1,
    const float* __restrict__ g1, const float* __restrict__ be1,
    const float* __restrict__ m1, const float* __restrict__ v1,
    const float* __restrict__ wcw2,
    __bf16* __restrict__ wcat, float* __restrict__ biascat,
    __bf16* __restrict__ wcw1b, __bf16* __restrict__ wcw2b,
    float* __restrict__ bnp) {
  int tid = blockIdx.x * blockDim.x + threadIdx.x;
  int stride = gridDim.x * blockDim.x;
  for (int i = tid; i < MTOT_ * CIN_; i += stride) {
    int row = i / CIN_, c = i & (CIN_ - 1);
    float v;
    if (row < 64)       v = w1[row * CIN_ + c];
    else if (row < 128) v = w2[(row - 64) * CIN_ + c];
    else                v = w3[(row - 128) * CIN_ + c];
    wcat[i] = f2bf(v);
  }
  for (int i = tid; i < MTOT_; i += stride) {
    float v;
    if (i < 64)       v = b1[i];
    else if (i < 128) v = b2[i - 64];
    else              v = b3[i - 128];
    biascat[i] = v;
  }
  for (int i = tid; i < 64 * CP_; i += stride) {
    int o = i / CP_, c = i % CP_;
    wcw1b[i] = (c < 66) ? f2bf(wcw1[o * 66 + c]) : f2bf(0.0f);
  }
  for (int i = tid; i < 64 * 64; i += stride) wcw2b[i] = f2bf(wcw2[i]);
  for (int i = tid; i < 66; i += stride) {
    float s = g0[i] * rsqrtf(v0[i] + 1e-5f);
    bnp[i] = s; bnp[66 + i] = be0[i] - m0[i] * s;
  }
  for (int i = tid; i < 64; i += stride) {
    float s = g1[i] * rsqrtf(v1[i] + 1e-5f);
    bnp[132 + i] = s; bnp[196 + i] = be1[i] - m1[i] * s;
  }
}

// ---------------------------------------------------------------------------
// Projection GEMM: proj[b][m][n] = sum_c wcat[m][c]*x[b][c][n] + biascat[m]
// M=640, K=512, N=HW per batch. One 16x16 tile per wave, 16 WMMA K-steps.
// ---------------------------------------------------------------------------
__global__ __launch_bounds__(256) void proj_gemm(const __bf16* __restrict__ xbT,
                                                 const __bf16* __restrict__ wcat,
                                                 const float* __restrict__ biascat,
                                                 float* __restrict__ proj) {
  const int wave = threadIdx.x >> 5;
  const int lane = threadIdx.x & 31;
  const int lm = lane & 15;
  const bool hi = lane >= 16;
  int tile = blockIdx.x * 8 + wave;                 // 4 * 40 * 64 = 10240 tiles
  if (tile >= 4 * 40 * 64) return;
  int b = tile / (40 * 64);
  int r = tile % (40 * 64);
  int m0 = (r / 64) * 16, n0 = (r % 64) * 16;

  v8f acc = {0.f, 0.f, 0.f, 0.f, 0.f, 0.f, 0.f, 0.f};
  const __bf16* arow = wcat + (size_t)(m0 + lm) * CIN_ + (hi ? 16 : 0);
  const __bf16* bcol = xbT + ((size_t)b * HW_ + (n0 + lm)) * CIN_ + (hi ? 16 : 0);
  for (int k0 = 0; k0 < CIN_; k0 += 32) {
    __builtin_prefetch(bcol + k0 + 128, 0, 0);      // -> global_prefetch_b8
    const v8bf* ap = (const v8bf*)(arow + k0);
    v16bf a = __builtin_shufflevector(ap[0], ap[1],
        0, 1, 2, 3, 4, 5, 6, 7, 8, 9, 10, 11, 12, 13, 14, 15);
    v16bf bb = *(const v16bf*)(bcol + k0);
    acc = __builtin_amdgcn_wmma_f32_16x16x32_bf16(false, a, false, bb,
                                                  (short)0, acc, false, false);
  }
  const int mbase = m0 + (hi ? 8 : 0);
  const int n = n0 + lm;
#pragma unroll
  for (int rr = 0; rr < 8; ++rr) {
    int m = mbase + rr;
    proj[((size_t)b * MTOT_ + m) * HW_ + n] = acc[rr] + biascat[m];
  }
}

// ---------------------------------------------------------------------------
// Fused attention: per (batch, 8-pixel tile)
//   phase0: async-copy stage weights (wcw1b/wcw2b) global->LDS (ASYNCcnt)
//   phase1: build tT[400 cols][96 ch] bf16 in LDS (subtraction + BN0 + ReLU)
//           (overlaps with the async weight DMA)
//   phase2: stage1 GEMM (A from LDS) -> BN1+ReLU -> t2T[400][64] bf16 LDS
//   phase3: stage2 GEMM (A from LDS) + bias -> wgt[400][64] f32 LDS
//   phase4: softmax over 49 window slots
//   phase5: grouped aggregation with x3 (proj rows 128..639) -> out
// LDS: wgt(aliases tT) 108800 + t2T 57600 + cw1L 12288 + cw2L 8192 = 186880 B
// ---------------------------------------------------------------------------
__global__ __launch_bounds__(256) void san_fused(const float* __restrict__ proj,
                                                 const __bf16* __restrict__ wcw1b,
                                                 const __bf16* __restrict__ wcw2b,
                                                 const float* __restrict__ bnp,
                                                 const float* __restrict__ wp,
                                                 const float* __restrict__ bp,
                                                 const float* __restrict__ bcw2,
                                                 float* __restrict__ out) {
  extern __shared__ char smem[];
  __bf16* tT   = (__bf16*)smem;                 // [NCOL_][TT_STRIDE]
  float*  wgt  = (float*)smem;                  // [NCOL_][WGT_STRIDE] (aliases tT)
  __bf16* t2T  = (__bf16*)(smem + SM_T2_OFF);   // [NCOL_][T2_STRIDE]
  __bf16* cw1L = (__bf16*)(smem + SM_CW1_OFF);  // [64][CP_]
  __bf16* cw2L = (__bf16*)(smem + SM_CW2_OFF);  // [64][64]

  const int tid = threadIdx.x;
  const int wave = tid >> 5;
  const int lane = tid & 31;
  const int lm = lane & 15;
  const bool hi = lane >= 16;
  const int b = blockIdx.x >> 7;               // 128 pixel-tiles per batch
  const int pixbase = (blockIdx.x & 127) * PIX_;

  // ---- phase 0: async DMA of stage weights into LDS (16B chunks) ----------
  // 64*96*2/16 = 768 chunks for cw1L, 64*64*2/16 = 512 chunks for cw2L.
  {
    const unsigned cw1base = (unsigned)(size_t)cw1L;
    const unsigned cw2base = (unsigned)(size_t)cw2L;
    for (int i = tid; i < 768 + 512; i += 256) {
      const char* src;
      unsigned dst;
      if (i < 768) { src = (const char*)wcw1b + i * 16; dst = cw1base + i * 16; }
      else { src = (const char*)wcw2b + (i - 768) * 16; dst = cw2base + (i - 768) * 16; }
      asm volatile("global_load_async_to_lds_b128 %0, %1, off"
                   :: "v"(dst), "v"(src) : "memory");
    }
  }

  const float wp00 = wp[0], wp01 = wp[1], wp10 = wp[2], wp11 = wp[3];
  const float bp0 = bp[0], bp1 = bp[1];

  // ---- phase 1: fill tT (zero-padded channels 66..95 and cols 392..399) ----
  for (int idx = tid; idx < NCOL_ * CP_; idx += 256) {
    int col = idx / CP_, cp = idx % CP_;
    float v = 0.f;
    if (col < KK_ * PIX_ && cp < 66) {
      int kk = col >> 3, j = col & 7;
      int n = pixbase + j, y = n >> 5, xq = n & 31;
      int yy = refl32(y + kk / 7 - 3), xx = refl32(xq + kk % 7 - 3);
      float center, neigh;
      if (cp < 64) {
        center = proj[((size_t)b * MTOT_ + cp) * HW_ + n];
        neigh  = proj[((size_t)b * MTOT_ + 64 + cp) * HW_ + yy * 32 + xx];
      } else {
        float lwc = -1.f + (2.f / 31.f) * (float)xq;
        float lhc = -1.f + (2.f / 31.f) * (float)y;
        float lwn = -1.f + (2.f / 31.f) * (float)xx;
        float lhn = -1.f + (2.f / 31.f) * (float)yy;
        if (cp == 64) { center = wp00 * lwc + wp01 * lhc + bp0;
                        neigh  = wp00 * lwn + wp01 * lhn + bp0; }
        else          { center = wp10 * lwc + wp11 * lhc + bp1;
                        neigh  = wp10 * lwn + wp11 * lhn + bp1; }
      }
      v = (center - neigh) * bnp[cp] + bnp[66 + cp];
      v = v > 0.f ? v : 0.f;
    }
    tT[col * TT_STRIDE + cp] = f2bf(v);
  }
  // wait for this wave's async weight DMA, then barrier makes LDS coherent
  asm volatile("s_wait_asynccnt 0x0" ::: "memory");
  __syncthreads();

  // ---- phase 2: stage1 GEMM 64x96 @ [96 x 400], BN1+ReLU -> t2T bf16 ----
  for (int tt = wave; tt < 100; tt += 8) {       // 4 M-tiles x 25 N-tiles
    int m0 = (tt / 25) * 16, n0 = (tt % 25) * 16;
    v8f acc = {0.f, 0.f, 0.f, 0.f, 0.f, 0.f, 0.f, 0.f};
    const __bf16* arow = cw1L + (m0 + lm) * CP_ + (hi ? 16 : 0);
    const __bf16* bcol = tT + (n0 + lm) * TT_STRIDE + (hi ? 16 : 0);
#pragma unroll
    for (int kk = 0; kk < 3; ++kk) {
      const v8bf* ap = (const v8bf*)(arow + kk * 32);
      v16bf a = __builtin_shufflevector(ap[0], ap[1],
          0, 1, 2, 3, 4, 5, 6, 7, 8, 9, 10, 11, 12, 13, 14, 15);
      v16bf bb = *(const v16bf*)(bcol + kk * 32);
      acc = __builtin_amdgcn_wmma_f32_16x16x32_bf16(false, a, false, bb,
                                                    (short)0, acc, false, false);
    }
    int mbase = m0 + (hi ? 8 : 0);
    v8bf ov;
#pragma unroll
    for (int rr = 0; rr < 8; ++rr) {
      int m = mbase + rr;
      float v = acc[rr] * bnp[132 + m] + bnp[196 + m];
      ov[rr] = f2bf(v > 0.f ? v : 0.f);
    }
    *(v8bf*)(t2T + (n0 + lm) * T2_STRIDE + mbase) = ov;
  }
  __syncthreads();

  // ---- phase 3: stage2 GEMM 64x64 @ [64 x 400] + bias -> wgt f32 ----
  for (int tt = wave; tt < 100; tt += 8) {
    int m0 = (tt / 25) * 16, n0 = (tt % 25) * 16;
    v8f acc = {0.f, 0.f, 0.f, 0.f, 0.f, 0.f, 0.f, 0.f};
    const __bf16* arow = cw2L + (m0 + lm) * 64 + (hi ? 16 : 0);
    const __bf16* bcol = t2T + (n0 + lm) * T2_STRIDE + (hi ? 16 : 0);
#pragma unroll
    for (int kk = 0; kk < 2; ++kk) {
      const v8bf* ap = (const v8bf*)(arow + kk * 32);
      v16bf a = __builtin_shufflevector(ap[0], ap[1],
          0, 1, 2, 3, 4, 5, 6, 7, 8, 9, 10, 11, 12, 13, 14, 15);
      v16bf bb = *(const v16bf*)(bcol + kk * 32);
      acc = __builtin_amdgcn_wmma_f32_16x16x32_bf16(false, a, false, bb,
                                                    (short)0, acc, false, false);
    }
    int mbase = m0 + (hi ? 8 : 0);
#pragma unroll
    for (int rr = 0; rr < 8; ++rr) {
      int m = mbase + rr;
      wgt[(n0 + lm) * WGT_STRIDE + m] = acc[rr] + bcw2[m];
    }
  }
  __syncthreads();

  // ---- phase 4: softmax over the 49 window offsets, per (group o, pixel j) ----
  for (int pp = tid; pp < 64 * PIX_; pp += 256) {
    int o = pp >> 3, j = pp & 7;
    float mx = -1e30f;
    for (int kk = 0; kk < KK_; ++kk)
      mx = fmaxf(mx, wgt[(kk * 8 + j) * WGT_STRIDE + o]);
    float s = 0.f;
    for (int kk = 0; kk < KK_; ++kk) {
      float e = __expf(wgt[(kk * 8 + j) * WGT_STRIDE + o] - mx);
      wgt[(kk * 8 + j) * WGT_STRIDE + o] = e;
      s += e;
    }
    float inv = 1.f / s;
    for (int kk = 0; kk < KK_; ++kk)
      wgt[(kk * 8 + j) * WGT_STRIDE + o] *= inv;
  }
  __syncthreads();

  // ---- phase 5: aggregation out[b,c,n] = sum_k wgt[c/8,k,n]*x3[b,c,nb(k,n)] ----
  for (int idx = tid; idx < 512 * PIX_; idx += 256) {
    int c = idx & 511, j = idx >> 9;
    int n = pixbase + j, y = n >> 5, xq = n & 31, g = c >> 3;
    const float* x3 = proj + ((size_t)b * MTOT_ + 128 + c) * HW_;
    float s = 0.f;
    for (int kk = 0; kk < KK_; ++kk) {
      int yy = refl32(y + kk / 7 - 3), xx = refl32(xq + kk % 7 - 3);
      s += wgt[(kk * 8 + j) * WGT_STRIDE + g] * x3[yy * 32 + xx];
    }
    out[((size_t)b * 512 + c) * HW_ + n] = s;
  }
}

// ---------------------------------------------------------------------------
extern "C" void kernel_launch(void* const* d_in, const int* in_sizes, int n_in,
                              void* d_out, int out_size, void* d_ws, size_t ws_size,
                              hipStream_t stream) {
  (void)in_sizes; (void)n_in; (void)out_size; (void)ws_size;
  const float* x    = (const float*)d_in[0];
  const float* w1   = (const float*)d_in[1];
  const float* b1   = (const float*)d_in[2];
  const float* w2   = (const float*)d_in[3];
  const float* b2   = (const float*)d_in[4];
  const float* w3   = (const float*)d_in[5];
  const float* b3   = (const float*)d_in[6];
  const float* wp   = (const float*)d_in[7];
  const float* bp   = (const float*)d_in[8];
  const float* g0   = (const float*)d_in[9];
  const float* be0  = (const float*)d_in[10];
  const float* m0   = (const float*)d_in[11];
  const float* v0   = (const float*)d_in[12];
  const float* wcw1 = (const float*)d_in[13];
  const float* g1   = (const float*)d_in[14];
  const float* be1  = (const float*)d_in[15];
  const float* m1   = (const float*)d_in[16];
  const float* v1   = (const float*)d_in[17];
  const float* wcw2 = (const float*)d_in[18];
  const float* bcw2 = (const float*)d_in[19];
  float* out = (float*)d_out;

  char* w = (char*)d_ws;
  size_t off = 0;
  auto alloc = [&](size_t bytes) {
    off = (off + 255) & ~(size_t)255;
    size_t o = off; off += bytes; return o;
  };
  __bf16* xbT     = (__bf16*)(w + alloc((size_t)4 * HW_ * CIN_ * 2));
  __bf16* wcat    = (__bf16*)(w + alloc((size_t)MTOT_ * CIN_ * 2));
  float*  biascat = (float*)(w + alloc((size_t)MTOT_ * 4));
  __bf16* wcw1b   = (__bf16*)(w + alloc((size_t)64 * CP_ * 2));
  __bf16* wcw2b   = (__bf16*)(w + alloc((size_t)64 * 64 * 2));
  float*  bnp     = (float*)(w + alloc((size_t)260 * 4));
  float*  proj    = (float*)(w + alloc((size_t)4 * MTOT_ * HW_ * 4));

  k_prep_x<<<(4 * HW_ * CIN_ + 255) / 256, 256, 0, stream>>>(x, xbT);
  k_prep_w<<<256, 256, 0, stream>>>(w1, b1, w2, b2, w3, b3, g0, be0, m0, v0,
                                    wcw1, g1, be1, m1, v1, wcw2,
                                    wcat, biascat, wcw1b, wcw2b, bnp);
  proj_gemm<<<1280, 256, 0, stream>>>(xbT, wcat, biascat, proj);

  const int smem_bytes = SM_TOTAL;  // 186880
  hipError_t err = hipFuncSetAttribute(
      (const void*)san_fused, hipFuncAttributeMaxDynamicSharedMemorySize,
      smem_bytes);
  (void)err;
  san_fused<<<512, 256, smem_bytes, stream>>>(proj, wcw1b, wcw2b, bnp,
                                              wp, bp, bcw2, out);
}
